// LinearMultiheadAttention_41205916238372
// MI455X (gfx1250) — matrix operand order repaired and checked by reading, not compiled, not run on hardware
//
#include <hip/hip_runtime.h>
#include <stdint.h>

// ---------------- problem constants ----------------
#define D_MODEL   1040
#define N_HEADS   16
#define FEAT      32
#define HEAD_DIM  65          // phi dim = 2*FEAT+1
#define BATCH     4
#define SEQ       4096
#define BS        (BATCH*SEQ)        // 16384 rows
#define PHS       96                 // padded per-head phi stride (mult of 32)
#define ROWSTR    (N_HEADS*PHS)      // 1536 bf16 per token row (phi_q layout)
#define TROWS     96                 // rows in transposed per-head layouts
#define KVD       96                 // kv K-dim (d), zero padded 80..95
#define KVE       80                 // kv e rows in kvT
#define KVSZ      (KVE*KVD)          // 7680 per (b,h)
#define EPS       1e-9f
#define NSPLIT    8                  // split-K factor for kv reduction

typedef __bf16 bf16_t;
typedef bf16_t v16bf __attribute__((ext_vector_type(16)));
typedef bf16_t v8bf  __attribute__((ext_vector_type(8)));
typedef float  v8f   __attribute__((ext_vector_type(8)));
typedef uint32_t u32x4 __attribute__((ext_vector_type(4)));
typedef int      i32x4 __attribute__((ext_vector_type(4)));
typedef int      i32x8 __attribute__((ext_vector_type(8)));

#if defined(__has_builtin)
#  if __has_builtin(__builtin_amdgcn_tensor_load_to_lds)
#    define HAVE_TDM 1
#  else
#    define HAVE_TDM 0
#  endif
#else
#  define HAVE_TDM 0
#endif

// ---------------- WMMA helpers ----------------
__device__ __forceinline__ v8f vzero8() {
  v8f z;
  #pragma unroll
  for (int i = 0; i < 8; ++i) z[i] = 0.0f;
  return z;
}

// A fragment (16x32 bf16, MxK): lane = row M=lane%16; half=lane/16:
// VGPR0-3 = K[half*8 .. +7], VGPR4-7 = K[16+half*8 .. +7]
__device__ __forceinline__ v16bf frag_a(const bf16_t* base, int stride, int lane) {
  const int r = lane & 15, hf = lane >> 4;
  const bf16_t* p = base + (size_t)r * stride + hf * 8;
  v8bf lo = *(const v8bf*)(p);
  v8bf hi = *(const v8bf*)(p + 16);
  return __builtin_shufflevector(lo, hi, 0,1,2,3,4,5,6,7,8,9,10,11,12,13,14,15);
}

// B fragment (32x16 bf16, KxN): lane = col N=lane%16; half=lane/16:
// VGPR0-3 = K[half*8 .. +7], VGPR4-7 = K[16+half*8 .. +7]. base points into a [n][k] tile.
__device__ __forceinline__ v16bf frag_b(const bf16_t* base, int stride, int lane) {
  const int c = lane & 15, hf = lane >> 4;
  const bf16_t* p = base + (size_t)c * stride + hf * 8;
  v8bf lo = *(const v8bf*)(p);
  v8bf hi = *(const v8bf*)(p + 16);
  return __builtin_shufflevector(lo, hi, 0,1,2,3,4,5,6,7,8,9,10,11,12,13,14,15);
}

__device__ __forceinline__ v8f wmma_bf16(v16bf a, v16bf b, v8f c) {
  return __builtin_amdgcn_wmma_f32_16x16x32_bf16(false, a, false, b, (short)0, c,
                                                 false, false);
}

// ---------------- TDM: async 2D bf16 tile load (rows x 32) into LDS ----------------
// Row payload is always 32 bf16 = 64B = 16 DWORDs; pad_interval=3 (16 DW),
// pad_amount=3 (4 DW = 16B) -> LDS row stride 80B = 40 bf16, matching frag loads.
// OOB in dim0/dim1 returns zero (handles K and N edges).
__device__ __forceinline__ void tdm_load_tile32(uint32_t lds_byte_addr,
                                                const bf16_t* gptr,
                                                uint32_t tensor_d0, uint32_t tensor_d1,
                                                uint32_t tile_d1,
                                                uint64_t row_stride_elems) {
#if HAVE_TDM
  uint64_t ga = (uint64_t)(uintptr_t)gptr;
  u32x4 g0 = { 1u,                                    // count=1 (valid, user mode)
               lds_byte_addr,                         // lds_addr
               (uint32_t)ga,                          // global_addr[31:0]
               (uint32_t)((ga >> 32) & 0x01FFFFFFu) | (2u << 30) };  // [56:32]|type=2
  const uint32_t tile_d0 = 32u;
  uint32_t d0 = (1u << 16) | (1u << 20) | (3u << 22) | (3u << 25);
  uint32_t d1 = (tensor_d0 & 0xFFFFu) << 16;
  uint32_t d2 = (tensor_d0 >> 16) | ((tensor_d1 & 0xFFFFu) << 16);
  uint32_t d3 = (tensor_d1 >> 16) | (tile_d0 << 16);
  uint32_t d4 = tile_d1;                              // tile_dim1; tile_dim2=0
  uint32_t d5 = (uint32_t)(row_stride_elems & 0xFFFFFFFFu);
  uint32_t d6 = (uint32_t)((row_stride_elems >> 32) & 0xFFFFu);   // stride1 = 0
  i32x8 g1 = { (int)d0, (int)d1, (int)d2, (int)d3, (int)d4, (int)d5, (int)d6, 0 };
  i32x4 gz = { 0, 0, 0, 0 };
#if __clang_major__ >= 23
  i32x8 gz8 = { 0, 0, 0, 0, 0, 0, 0, 0 };
  __builtin_amdgcn_tensor_load_to_lds(g0, g1, gz, gz, gz8, 0);
#else
  __builtin_amdgcn_tensor_load_to_lds(g0, g1, gz, gz, 0);
#endif
#endif
}

// Fallback manual staging (used only if TDM builtin is unavailable)
__device__ __forceinline__ void stage_tile(bf16_t (*dst)[40], const bf16_t* src,
                                           int rows, int valid_rows,
                                           size_t row_stride, int kvalid,
                                           int tid, int nthreads) {
  for (int idx = tid; idx < rows * 4; idx += nthreads) {
    int r = idx >> 2, kg = (idx & 3) << 3;
    bf16_t tmp[8];
    if (r < valid_rows && kg + 8 <= kvalid) {
      *(v8bf*)tmp = *(const v8bf*)(src + (size_t)r * row_stride + kg);
    } else {
      for (int j = 0; j < 8; ++j)
        tmp[j] = (r < valid_rows && kg + j < kvalid)
                     ? src[(size_t)r * row_stride + kg + j] : (bf16_t)0.0f;
    }
    *(v8bf*)&dst[r][kg] = *(v8bf*)tmp;
  }
}

// ---------------- small utility kernels ----------------
__global__ void cast_f32_bf16(const float* __restrict__ s, bf16_t* __restrict__ d, int n) {
  int i = blockIdx.x * blockDim.x + threadIdx.x;
  int stride = gridDim.x * blockDim.x;
  for (; i < n; i += stride) d[i] = (bf16_t)s[i];
}

// W[K][N] -> WT[N][K], f32 -> bf16
__global__ void cast_transpose(const float* __restrict__ W, bf16_t* __restrict__ WT,
                               int K, int N) {
  int i = blockIdx.x * blockDim.x + threadIdx.x;
  if (i >= K * N) return;
  int k = i / N, n = i % N;
  WT[(size_t)n * K + k] = (bf16_t)W[i];
}

__global__ void zero_f32(float* __restrict__ p, int n) {
  int i = blockIdx.x * blockDim.x + threadIdx.x;
  if (i < n) p[i] = 0.0f;
}

// phi_q pads: [m][h*96]: d=0 -> 1, d=65..95 -> 0
__global__ void init_phiq(bf16_t* __restrict__ phiq) {
  int i = blockIdx.x * blockDim.x + threadIdx.x;
  if (i >= BS * N_HEADS) return;
  int m = i >> 4, h = i & 15;
  size_t base = (size_t)m * ROWSTR + h * PHS;
  phiq[base] = (bf16_t)1.0f;
  for (int d = HEAD_DIM; d < PHS; ++d) phiq[base + d] = (bf16_t)0.0f;
}

// transposed pads: phi_kT row0 = 1, rows 65..79 = 0; vT row 65 = 1, rows 66..79 = 0
__global__ void init_T(bf16_t* __restrict__ phikT, bf16_t* __restrict__ vT) {
  int i = blockIdx.x * blockDim.x + threadIdx.x;
  if (i >= BATCH * N_HEADS * SEQ) return;
  int bh = i >> 12, n = i & (SEQ - 1);
  size_t base = (size_t)bh * TROWS * SEQ;
  phikT[base + n] = (bf16_t)1.0f;
  for (int r = HEAD_DIM; r < 80; ++r) phikT[base + (size_t)r * SEQ + n] = (bf16_t)0.0f;
  vT[base + (size_t)HEAD_DIM * SEQ + n] = (bf16_t)1.0f;
  for (int r = HEAD_DIM + 1; r < 80; ++r) vT[base + (size_t)r * SEQ + n] = (bf16_t)0.0f;
}

__global__ void qsum_kernel(const bf16_t* __restrict__ phiq, float* __restrict__ qsum) {
  int i = blockIdx.x * blockDim.x + threadIdx.x;
  if (i >= BATCH * N_HEADS * SEQ) return;
  int bh = i >> 12, s = i & (SEQ - 1);
  int b = bh >> 4, h = bh & 15;
  size_t base = (size_t)(b * SEQ + s) * ROWSTR + h * PHS;
  float acc = 0.0f;
  for (int d = 0; d < HEAD_DIM; ++d) acc += (float)phiq[base + d];
  qsum[i] = acc;
}

__global__ void kv_cast(const float* __restrict__ kvf, bf16_t* __restrict__ kvT, int n) {
  int i = blockIdx.x * blockDim.x + threadIdx.x;
  if (i < n) kvT[i] = (bf16_t)kvf[i];
}

// ---------------- generic tiled WMMA GEMM: C[M,N] = A[M,K] * BT[N,K]^T ----------------
// A row-major [M][K] bf16; BT row-major [N][K] bf16 (pre-transposed weights).
// Block tile 128x128, BK=32, 8 waves (2M x 4N), wave computes 64x32.
// Double-buffered LDS: wave0 drives TDM; the DMA for step ks+1 overlaps the WMMA
// compute of step ks.  s_wait_tensorcnt(0) at iteration start only covers the tile
// prefetched a full compute-phase earlier.
// MODE 0: f32 C.  MODE 1: phi_q epilogue.  MODE 3: phi_kT epilogue.  MODE 4: vT.
template <int MODE>
__global__ __launch_bounds__(256)
void gemm_bf16_wmma(const bf16_t* __restrict__ A, const bf16_t* __restrict__ BT,
                    int N, int K, float* __restrict__ cout, bf16_t* __restrict__ aux) {
  __shared__ alignas(16) bf16_t As[2][128][40];   // [buf][m][k]
  __shared__ alignas(16) bf16_t Bs[2][128][40];   // [buf][n][k]
  const int tid = threadIdx.x, lane = tid & 31, w = tid >> 5;
  const int n0 = blockIdx.x * 128;
  const int m0 = blockIdx.y * 128;
  const int mB = (w & 1) * 64;
  const int nB = (w >> 1) * 32;
  const int ksteps = (K + 31) >> 5;

#if HAVE_TDM
  const uint32_t ldsA[2] = { (uint32_t)(uintptr_t)&As[0][0][0],
                             (uint32_t)(uintptr_t)&As[1][0][0] };
  const uint32_t ldsB[2] = { (uint32_t)(uintptr_t)&Bs[0][0][0],
                             (uint32_t)(uintptr_t)&Bs[1][0][0] };
  if (w == 0) {  // prologue: prefetch tile 0
    tdm_load_tile32(ldsA[0], A + (size_t)m0 * K, (uint32_t)K, 128u, 128u, (uint64_t)K);
    tdm_load_tile32(ldsB[0], BT + (size_t)n0 * K, (uint32_t)K, (uint32_t)(N - n0),
                    128u, (uint64_t)K);
  }
#endif

  v8f acc[4][2];
  #pragma unroll
  for (int i = 0; i < 4; ++i)
    #pragma unroll
    for (int j = 0; j < 2; ++j) acc[i][j] = vzero8();

  for (int ks = 0; ks < ksteps; ++ks) {
    const int cur = ks & 1;
#if HAVE_TDM
    if (w == 0) {
      __builtin_amdgcn_s_wait_tensorcnt(0);     // tile ks resident
      if (ks + 1 < ksteps) {                    // prefetch tile ks+1 (overlaps compute)
        const int kn = (ks + 1) << 5;
        tdm_load_tile32(ldsA[cur ^ 1], A + (size_t)m0 * K + kn, (uint32_t)(K - kn),
                        128u, 128u, (uint64_t)K);
        tdm_load_tile32(ldsB[cur ^ 1], BT + (size_t)n0 * K + kn, (uint32_t)(K - kn),
                        (uint32_t)(N - n0), 128u, (uint64_t)K);
      }
    }
#else
    {
      const int kb = ks << 5;
      stage_tile(As[cur], A + (size_t)m0 * K + kb, 128, 128, (size_t)K, K - kb, tid, 256);
      stage_tile(Bs[cur], BT + (size_t)n0 * K + kb, 128,
                 (N - n0 < 128 ? N - n0 : 128), (size_t)K, K - kb, tid, 256);
    }
#endif
    __syncthreads();
    v16bf af[4], bfr[2];
    #pragma unroll
    for (int i = 0; i < 4; ++i) af[i] = frag_a(&As[cur][mB + 16 * i][0], 40, lane);
    #pragma unroll
    for (int j = 0; j < 2; ++j) bfr[j] = frag_b(&Bs[cur][nB + 16 * j][0], 40, lane);
    #pragma unroll
    for (int i = 0; i < 4; ++i)
      #pragma unroll
      for (int j = 0; j < 2; ++j) acc[i][j] = wmma_bf16(af[i], bfr[j], acc[i][j]);
    __syncthreads();
  }

  // epilogue: element (r + 8*half, lane%16) of each 16x16 tile
  const int cl = lane & 15, hf = lane >> 4;
  #pragma unroll
  for (int i = 0; i < 4; ++i) {
    #pragma unroll
    for (int j = 0; j < 2; ++j) {
      int mt = m0 + mB + 16 * i, nt = n0 + nB + 16 * j;
      #pragma unroll
      for (int r = 0; r < 8; ++r) {
        int row = mt + r + 8 * hf;
        int col = nt + cl;
        float v = acc[i][j][r];
        if (MODE == 0) {
          if (col < N) cout[(size_t)row * N + col] = v;
        } else if (MODE == 1) {         // phi_q: [m][h*96 + {1+f, 33+f}]
          int h = col >> 5, f = col & 31;
          size_t base = (size_t)row * ROWSTR + h * PHS;
          aux[base + 1 + f] = (bf16_t)v;
          aux[base + 1 + FEAT + f] = (bf16_t)(0.5f * v * v);
        } else if (MODE == 3) {         // phi_kT: [bh][d][s]
          int b = row >> 12, s = row & (SEQ - 1);
          int h = col >> 5, f = col & 31;
          size_t base = (size_t)(b * N_HEADS + h) * TROWS * SEQ;
          aux[base + (size_t)(1 + f) * SEQ + s] = (bf16_t)v;
          aux[base + (size_t)(1 + FEAT + f) * SEQ + s] = (bf16_t)(0.5f * v * v);
        } else {                        // MODE 4, vT: [bh][e][s]
          if (col < N) {
            int b = row >> 12, s = row & (SEQ - 1);
            int h = col / HEAD_DIM, e = col % HEAD_DIM;
            aux[(size_t)((b * N_HEADS + h) * TROWS + e) * SEQ + s] = (bf16_t)v;
          }
        }
      }
    }
  }
}

// ---------------- kv = phi(k)^T @ v_pad per (b,h), split-K over S ----------------
// grid (64, NSPLIT), 5 waves. A = phi_kT rows d (80), B = vT rows e (80), K = n.
// Double-buffered TDM pipeline; partials accumulated into f32 kvT-layout
// [bh][e][96 d] via global atomics.
__global__ __launch_bounds__(160)
void kv_wmma(const bf16_t* __restrict__ phikT, const bf16_t* __restrict__ vT,
             float* __restrict__ kvf) {
  __shared__ alignas(16) bf16_t Ks[2][80][40];  // [buf][d][n_local]
  __shared__ alignas(16) bf16_t Vs[2][80][40];  // [buf][e][n_local]
  const int tid = threadIdx.x, lane = tid & 31, w = tid >> 5;
  const int bh = blockIdx.x;
  const int nbase = blockIdx.y * (SEQ / NSPLIT);
  const size_t gbase = (size_t)bh * TROWS * SEQ;
  const int steps = (SEQ / NSPLIT) / 32;        // 16

#if HAVE_TDM
  const uint32_t ldsK[2] = { (uint32_t)(uintptr_t)&Ks[0][0][0],
                             (uint32_t)(uintptr_t)&Ks[1][0][0] };
  const uint32_t ldsV[2] = { (uint32_t)(uintptr_t)&Vs[0][0][0],
                             (uint32_t)(uintptr_t)&Vs[1][0][0] };
  if (w == 0) {
    tdm_load_tile32(ldsK[0], phikT + gbase + nbase, (uint32_t)(SEQ - nbase), 80u, 80u,
                    (uint64_t)SEQ);
    tdm_load_tile32(ldsV[0], vT + gbase + nbase, (uint32_t)(SEQ - nbase), 80u, 80u,
                    (uint64_t)SEQ);
  }
#endif

  v8f acc[5];
  #pragma unroll
  for (int t = 0; t < 5; ++t) acc[t] = vzero8();

  for (int it = 0; it < steps; ++it) {
    const int cur = it & 1;
#if HAVE_TDM
    if (w == 0) {
      __builtin_amdgcn_s_wait_tensorcnt(0);
      if (it + 1 < steps) {
        const int nn = nbase + (it + 1) * 32;
        tdm_load_tile32(ldsK[cur ^ 1], phikT + gbase + nn, (uint32_t)(SEQ - nn), 80u,
                        80u, (uint64_t)SEQ);
        tdm_load_tile32(ldsV[cur ^ 1], vT + gbase + nn, (uint32_t)(SEQ - nn), 80u,
                        80u, (uint64_t)SEQ);
      }
    }
#else
    {
      const int nb = nbase + it * 32;
      stage_tile(Ks[cur], phikT + gbase + nb, 80, 80, (size_t)SEQ, 32, tid, 160);
      stage_tile(Vs[cur], vT + gbase + nb, 80, 80, (size_t)SEQ, 32, tid, 160);
    }
#endif
    __syncthreads();
    v16bf a = frag_a(&Ks[cur][w * 16][0], 40, lane);
    #pragma unroll
    for (int t = 0; t < 5; ++t) {
      v16bf bb = frag_b(&Vs[cur][t * 16][0], 40, lane);
      acc[t] = wmma_bf16(a, bb, acc[t]);
    }
    __syncthreads();
  }

  const int cl = lane & 15, hf = lane >> 4;
  #pragma unroll
  for (int t = 0; t < 5; ++t)
    #pragma unroll
    for (int r = 0; r < 8; ++r) {
      int d = w * 16 + r + 8 * hf;   // kv row (K-dim of qkv)
      int e = t * 16 + cl;           // kv col (N-dim of qkv)
      atomicAdd(&kvf[(size_t)bh * KVSZ + (size_t)e * KVD + d], acc[t][r]);
    }
}

// ---------------- qkv = phi(q) @ kv, normalize, mask -> ctx (bf16) ----------------
// grid (S/128, 64), 8 waves x 16 rows. K = 96 (3 steps). kvT rows are contiguous
// in d so staging is pure vector copies; row 65 of kvT is ksum.
__global__ __launch_bounds__(256)
void qkv_wmma(const bf16_t* __restrict__ phiq, const bf16_t* __restrict__ kvT,
              const float* __restrict__ qsum, const int* __restrict__ mask,
              bf16_t* __restrict__ ctx) {
  __shared__ alignas(16) bf16_t Kvs[80][104];  // [e][d]
  const int tid = threadIdx.x, lane = tid & 31, w = tid >> 5;
  const int m0 = blockIdx.x * 128;
  const int bh = blockIdx.y;
  const int b = bh >> 4, h = bh & 15;

  for (int idx = tid; idx < 960; idx += 256) {
    int e = idx / 12, dg = (idx % 12) * 8;
    v8bf t = *(const v8bf*)(kvT + (size_t)bh * KVSZ + (size_t)e * KVD + dg);
    *(v8bf*)&Kvs[e][dg] = t;
  }
  __syncthreads();

  v8f acc[5];
  #pragma unroll
  for (int t = 0; t < 5; ++t) acc[t] = vzero8();

  const size_t arow = (size_t)(b * SEQ + m0 + w * 16) * ROWSTR + h * PHS;
  #pragma unroll
  for (int ks = 0; ks < 3; ++ks) {
    v16bf a = frag_a(phiq + arow + ks * 32, ROWSTR, lane);
    #pragma unroll
    for (int t = 0; t < 5; ++t) {
      v16bf bb = frag_b(&Kvs[t * 16][ks * 32], 104, lane);
      acc[t] = wmma_bf16(a, bb, acc[t]);
    }
  }

  const int cl = lane & 15, hf = lane >> 4;
  #pragma unroll
  for (int r = 0; r < 8; ++r) {
    int s = m0 + w * 16 + r + 8 * hf;
    float qs = qsum[bh * SEQ + s];
    int mk = mask[b * SEQ + s];
    #pragma unroll
    for (int t = 0; t < 5; ++t) {
      int e = t * 16 + cl;
      if (e < HEAD_DIM) {
        float ksv = (float)kvT[(size_t)bh * KVSZ + (size_t)HEAD_DIM * KVD + e];
        float v = acc[t][r] / (qs * ksv + EPS);
        if (mk == 0) v = 0.0f;
        ctx[(size_t)(b * SEQ + s) * D_MODEL + h * HEAD_DIM + e] = (bf16_t)v;
      }
    }
  }
}

// ---------------- host side ----------------
extern "C" void kernel_launch(void* const* d_in, const int* in_sizes, int n_in,
                              void* d_out, int out_size, void* d_ws, size_t ws_size,
                              hipStream_t stream) {
  const float* hidden = (const float*)d_in[0];
  const int*   mask   = (const int*)d_in[1];
  const float* Wq     = (const float*)d_in[2];
  const float* Wk     = (const float*)d_in[3];
  const float* Wv     = (const float*)d_in[4];
  const float* Wo     = (const float*)d_in[5];
  float* out = (float*)d_out;

  size_t off = 0;
  auto alloc = [&](size_t nbytes) -> void* {
    off = (off + 255) & ~(size_t)255;
    void* p = (char*)d_ws + off;
    off += nbytes;
    return p;
  };
  bf16_t* hb    = (bf16_t*)alloc((size_t)BS * D_MODEL * 2);
  bf16_t* wqT   = (bf16_t*)alloc((size_t)D_MODEL * N_HEADS * FEAT * 2);
  bf16_t* wkT   = (bf16_t*)alloc((size_t)D_MODEL * N_HEADS * FEAT * 2);
  bf16_t* wvT   = (bf16_t*)alloc((size_t)D_MODEL * D_MODEL * 2);
  bf16_t* woT   = (bf16_t*)alloc((size_t)D_MODEL * D_MODEL * 2);
  bf16_t* phiq  = (bf16_t*)alloc((size_t)BS * ROWSTR * 2);
  bf16_t* phikT = (bf16_t*)alloc((size_t)BATCH * N_HEADS * TROWS * SEQ * 2);
  bf16_t* vT    = (bf16_t*)alloc((size_t)BATCH * N_HEADS * TROWS * SEQ * 2);
  float*  kvf   = (float*)alloc((size_t)BATCH * N_HEADS * KVSZ * 4);
  bf16_t* kvT   = (bf16_t*)alloc((size_t)BATCH * N_HEADS * KVSZ * 2);
  float*  qsm   = (float*)alloc((size_t)BATCH * N_HEADS * SEQ * 4);
  bf16_t* ctx   = (bf16_t*)alloc((size_t)BS * D_MODEL * 2);

  // 1) casts, weight transposes, pad init, kv zero
  cast_f32_bf16<<<2048, 256, 0, stream>>>(hidden, hb, BS * D_MODEL);
  cast_transpose<<<(D_MODEL * 512 + 255) / 256, 256, 0, stream>>>(Wq, wqT, D_MODEL, 512);
  cast_transpose<<<(D_MODEL * 512 + 255) / 256, 256, 0, stream>>>(Wk, wkT, D_MODEL, 512);
  cast_transpose<<<(D_MODEL * D_MODEL + 255) / 256, 256, 0, stream>>>(Wv, wvT, D_MODEL, D_MODEL);
  cast_transpose<<<(D_MODEL * D_MODEL + 255) / 256, 256, 0, stream>>>(Wo, woT, D_MODEL, D_MODEL);
  init_phiq<<<(BS * N_HEADS + 255) / 256, 256, 0, stream>>>(phiq);
  init_T<<<(BATCH * N_HEADS * SEQ + 255) / 256, 256, 0, stream>>>(phikT, vT);
  zero_f32<<<(BATCH * N_HEADS * KVSZ + 255) / 256, 256, 0, stream>>>(
      kvf, BATCH * N_HEADS * KVSZ);

  // 2) projections (fused phi / transpose epilogues)
  gemm_bf16_wmma<1><<<dim3(4, BS / 128), 256, 0, stream>>>(
      hb, wqT, N_HEADS * FEAT, D_MODEL, nullptr, phiq);
  gemm_bf16_wmma<3><<<dim3(4, BS / 128), 256, 0, stream>>>(
      hb, wkT, N_HEADS * FEAT, D_MODEL, nullptr, phikT);
  gemm_bf16_wmma<4><<<dim3(9, BS / 128), 256, 0, stream>>>(
      hb, wvT, D_MODEL, D_MODEL, nullptr, vT);

  // 3) normalizers + kv state (split-K atomics), then cast to bf16
  qsum_kernel<<<(BATCH * N_HEADS * SEQ + 255) / 256, 256, 0, stream>>>(phiq, qsm);
  kv_wmma<<<dim3(BATCH * N_HEADS, NSPLIT), 160, 0, stream>>>(phikT, vT, kvf);
  kv_cast<<<(BATCH * N_HEADS * KVSZ + 255) / 256, 256, 0, stream>>>(
      kvf, kvT, BATCH * N_HEADS * KVSZ);

  // 4) qkv + normalize + mask -> ctx
  qkv_wmma<<<dim3(SEQ / 128, BATCH * N_HEADS), 256, 0, stream>>>(
      phiq, kvT, qsm, mask, ctx);

  // 5) output projection
  gemm_bf16_wmma<0><<<dim3(9, BS / 128), 256, 0, stream>>>(
      ctx, woT, D_MODEL, D_MODEL, out, nullptr);
}
